// DEAL_8718783611446
// MI455X (gfx1250) — matrix operand order repaired and checked by reading, not compiled, not run on hardware
//
#include <hip/hip_runtime.h>

// ---------------------------------------------------------------------------
// MI455X (gfx1250) CG spline-denoiser.
//
// ~830 GFLOP total, >90% in three conv families (8->128 9x9, adjoint
// 128->8 9x9, 128->128 3x3) -> implicit GEMM on v_wmma_f32_16x16x32_bf16
// (bf16 A/B, fp32 accumulate).  B fragments come from LDS as two aligned
// 16B loads per WMMA (channel-minor patch for CIN=128, true im2col tile for
// CIN=8); A fragments are two 16B global loads from L2-resident packed
// weights (prefetched).  Cheap work (1->4/4->8 convs, splines, CG updates)
// stays fp32 VALU.
// ---------------------------------------------------------------------------

typedef __bf16 bf16_t;
typedef __attribute__((ext_vector_type(16))) __bf16 v16bf;
typedef __attribute__((ext_vector_type(8)))  __bf16 v8bf;
typedef __attribute__((ext_vector_type(8)))  float  v8f;

#define NB   8
#define HDIM 128
#define WDIM 128
#define HW   (HDIM*WDIM)
#define LCH  128

__device__ __forceinline__ float spline_eval(float x, const float* __restrict__ c,
                                             int K, float xmin, float xmax)
{
  float step = (xmax - xmin) / (float)(K - 1);
  float t = (x - xmin) / step;
  float f = floorf(t);
  f = fminf(fmaxf(f, 0.0f), (float)(K - 2));
  int idx = (int)f;
  float frac = t - f;
  return c[idx] * (1.0f - frac) + c[idx + 1] * frac;
}

// ---------------------------------------------------------------------------
// Weight packing: OIHW fp32 -> [co][r], zero padded to RPAD cols.
//   chmin=0 : r = ci*K*K + ky*K + kx   (channel-major, for im2col path)
//   chmin=1 : r = (ky*K + kx)*Ci + ci  (tap-major,    for channel-minor path)
//   flip=1  : adjoint weights (in/out channels swapped, taps flipped)
// ---------------------------------------------------------------------------
template<typename T>
__global__ void k_pack(const float* __restrict__ w, T* __restrict__ o,
                       int O, int I, int K, int RPAD, int flip, int chmin, int total)
{
  int idx = blockIdx.x * blockDim.x + threadIdx.x;
  if (idx >= total) return;
  int co = idx / RPAD;
  int r  = idx - co * RPAD;
  int KK = K * K;
  int CiR = flip ? O : I;
  float v = 0.0f;
  if (r < CiR * KK) {
    int ci, dy, dx;
    if (chmin) {
      ci = r % CiR;
      int tap = r / CiR;
      dy = tap / K; dx = tap - dy * K;
    } else {
      ci = r / KK;
      int rem = r - ci * KK;
      dy = rem / K; dx = rem - dy * K;
    }
    if (!flip) {
      if (co < O) v = w[(size_t)(co * I + ci) * KK + dy * K + dx];
    } else {
      if (co < I) v = w[(size_t)(ci * I + co) * KK + (K - 1 - dy) * K + (K - 1 - dx)];
    }
  }
  o[idx] = (T)v;
}

// ---------------------------------------------------------------------------
// WMMA implicit-GEMM conv (NCHW, stride 1, SAME padding).
//   M = cout tile (16), N = 16 pixels of one row, K = ci*KSZ*KSZ (padded 32).
// Block = 8 waves; one block per (x-tile, row, batch).
//   COUTP==128 : 8 waves = 8 cout tiles sharing the staged input.
//   COUTP==16  : 8 waves split the K reduction; wave 0 sums partials in LDS (buffer
//                aliases the patch after a barrier).
//   CHMIN==1   : LDS patch stored channel-minor; a 16-wide K segment is 16
//                consecutive channels at one tap -> contiguous 16B loads.
//   CHMIN==0   : LDS im2col tile [16][RPAD] built in a second staging phase.
// ---------------------------------------------------------------------------
template<int CIN, int COUTP, int COUTR, int KSZ, int CHMIN>
__global__ __launch_bounds__(256)
void k_conv_wmma(const float* __restrict__ in,
                 const bf16_t* __restrict__ wp,
                 float* __restrict__ out)
{
  constexpr int PAD    = KSZ / 2;
  constexpr int R      = CIN * KSZ * KSZ;
  constexpr int RPAD   = (R + 31) & ~31;
  constexpr int NCT    = COUTP / 16;
  constexpr int KWAVES = 8 / NCT;
  constexpr int PW     = 16 + KSZ - 1;
  constexpr int CPAD   = CIN + 8;                       // bank-decorrelated
  constexpr int SM_CH  = KSZ * PW * CPAD * 2;           // chmin patch bytes
  constexpr int SM_RED = (KWAVES > 1) ? 8 * 256 * 4 : 0;
  constexpr int SM_IM  = CIN * KSZ * PW * 2 + 16 * RPAD * 2;
  constexpr int SMEM   = CHMIN ? (SM_CH > SM_RED ? SM_CH : SM_RED) : SM_IM;
  __shared__ alignas(16) char smem[SMEM];

  const int x0 = blockIdx.x * 16;
  const int gy = blockIdx.y;
  const int b  = blockIdx.z;
  const float* inb = in + (size_t)b * CIN * HW;

  bf16_t* patchC = (bf16_t*)smem;                       // CHMIN: [yy][xx][ci]
  bf16_t* patchS = (bf16_t*)smem;                       // !CHMIN: [ci][yy][xx]
  bf16_t* im2    = (bf16_t*)(smem + CIN * KSZ * PW * 2);// !CHMIN: [n][RPAD]
  float*  red    = (float*)smem;                        // aliases patch (post-sync)

  // ---- stage input (global reads coalesced along x) -------------------
  for (int i = threadIdx.x; i < CIN * KSZ * PW; i += 256) {
    int ci  = i / (KSZ * PW);
    int rem = i - ci * (KSZ * PW);
    int yy  = rem / PW;
    int xx  = rem - yy * PW;
    int iy  = gy + yy - PAD;
    int ix  = x0 + xx - PAD;
    float v = 0.0f;
    if (iy >= 0 && iy < HDIM && ix >= 0 && ix < WDIM)
      v = inb[(size_t)ci * HW + iy * WDIM + ix];
    if constexpr (CHMIN) patchC[(yy * PW + xx) * CPAD + ci] = (bf16_t)v;
    else                 patchS[i] = (bf16_t)v;
  }
  __syncthreads();

  if constexpr (!CHMIN) {
    // ---- expand to im2col tile (zero-filled K padding) ----------------
    for (int j = threadIdx.x; j < 16 * RPAD; j += 256) {
      int n = j / RPAD;
      int r = j - n * RPAD;
      bf16_t v = (bf16_t)0.0f;
      if (r < R) {
        int ci  = r / (KSZ * KSZ);
        int tap = r - ci * (KSZ * KSZ);
        int dy  = tap / KSZ;
        int dx  = tap - dy * KSZ;
        v = patchS[ci * (KSZ * PW) + dy * PW + (n + dx)];
      }
      im2[j] = v;
    }
    __syncthreads();
  }

  const int wave  = threadIdx.x >> 5;
  const int lane  = threadIdx.x & 31;
  const int ct    = wave % NCT;
  const int kwave = wave / NCT;
  const int m0    = ct * 16;
  const int l15   = lane & 15;
  const int kg    = lane >> 4;

  v8f acc = {};
  const bf16_t* wrow = wp + (size_t)(m0 + l15) * RPAD + 16 * kg;
  __builtin_prefetch(wrow + kwave * 32, 0, 1);          // global_prefetch_b8

  auto step = [&](int kb) {
    // A fragment: lane M=l15; half i -> K = kb + 16*kg + i (contiguous).
    v8bf alo = *reinterpret_cast<const v8bf*>(wrow + kb);
    v8bf ahi = *reinterpret_cast<const v8bf*>(wrow + kb + 8);
    v16bf afrag;
#pragma unroll
    for (int i = 0; i < 8; ++i) { afrag[i] = alo[i]; afrag[i + 8] = ahi[i]; }

    // B fragment: lane N=l15; half i -> K = kb + 16*kg + i (contiguous).
    const int k16 = kb + 16 * kg;
    const bf16_t* bp;
    if constexpr (CHMIN) {
      int tap = k16 / CIN;
      int ci0 = k16 - tap * CIN;
      int dy  = tap / KSZ;
      int dx  = tap - dy * KSZ;
      bp = patchC + ((dy * PW) + (l15 + dx)) * CPAD + ci0;
    } else {
      bp = im2 + l15 * RPAD + k16;
    }
    v8bf b0 = *reinterpret_cast<const v8bf*>(bp);
    v8bf b1 = *reinterpret_cast<const v8bf*>(bp + 8);
    v16bf bfrag;
#pragma unroll
    for (int i = 0; i < 8; ++i) { bfrag[i] = b0[i]; bfrag[i + 8] = b1[i]; }

    acc = __builtin_amdgcn_wmma_f32_16x16x32_bf16(false, afrag, false, bfrag,
                                                  (short)0, acc, false, false);
  };

  if constexpr (KWAVES == 1) {
#pragma unroll
    for (int kb = 0; kb < RPAD; kb += 32) step(kb);
  } else {
#pragma unroll 2
    for (int kb = kwave * 32; kb < RPAD; kb += 32 * KWAVES) step(kb);
  }

  if constexpr (KWAVES > 1) {
    __syncthreads();                                    // patch reads done; alias as red
#pragma unroll
    for (int rr = 0; rr < 8; ++rr)
      red[wave * 256 + (8 * kg + rr) * 16 + l15] = acc[rr];
    __syncthreads();
    if (wave != 0) return;
#pragma unroll
    for (int rr = 0; rr < 8; ++rr) {
      float s = 0.0f;
#pragma unroll
      for (int w2 = 0; w2 < KWAVES; ++w2)
        s += red[w2 * 256 + (8 * kg + rr) * 16 + l15];
      acc[rr] = s;
    }
  }

  // C/D layout: lane = N, vgpr rr -> M = m0 + 8*kg + rr.
  float* outp = out + (size_t)b * COUTR * HW + (size_t)gy * WDIM + (x0 + l15);
#pragma unroll
  for (int rr = 0; rr < 8; ++rr) {
    int m = m0 + 8 * kg + rr;
    if (m < COUTR) outp[(size_t)m * HW] = acc[rr];
  }
}

// ---------------------------------------------------------------------------
// Small direct fp32 convs (1->4, 4->8 and adjoints) using packed weights.
// ---------------------------------------------------------------------------
__global__ void k_conv_small(const float* __restrict__ in, const float* __restrict__ wp,
                             float* __restrict__ out, int Ci, int Co, int K, int total)
{
  int idx = blockIdx.x * blockDim.x + threadIdx.x;
  if (idx >= total) return;
  int x  = idx & (WDIM - 1);
  int y  = (idx >> 7) & (HDIM - 1);
  int co = (idx >> 14) % Co;
  int b  = idx / (Co * HW);
  int pad = K >> 1;
  const float* wr  = wp + (size_t)co * Ci * K * K;
  const float* inb = in + (size_t)b * Ci * HW;
  float s = 0.0f;
  for (int ci = 0; ci < Ci; ++ci) {
    const float* inc = inb + (size_t)ci * HW;
    const float* wc  = wr + ci * K * K;
    for (int dy = 0; dy < K; ++dy) {
      int iy = y + dy - pad;
      if (iy < 0 || iy >= HDIM) continue;
      for (int dx = 0; dx < K; ++dx) {
        int ix = x + dx - pad;
        if (ix < 0 || ix >= WDIM) continue;
        s += inc[iy * WDIM + ix] * wc[dy * K + dx];
      }
    }
  }
  out[idx] = s;
}

// ---------------------------------------------------------------------------
// Elementwise / spline / CG kernels.
// ---------------------------------------------------------------------------
__global__ void k_lam_scal(const float* __restrict__ sigma, const float* __restrict__ c_lam,
                           const float* __restrict__ c_scal, float* __restrict__ lmbda,
                           float* __restrict__ inv1pl, float* __restrict__ scaling)
{
  int t = blockIdx.x * blockDim.x + threadIdx.x;
  if (t >= NB * LCH) return;
  int b = t >> 7, c = t & 127;
  float s = sigma[b];
  if (c == 0) {
    float l = spline_eval(s, c_lam, 53, -1.0f, 51.0f);
    lmbda[b]  = l;
    inv1pl[b] = 1.0f / (1.0f + l);
  }
  float sc = spline_eval(s, c_scal + c * 14, 14, -1.0f, 51.0f);
  scaling[t] = __expf(sc) / (s + 1e-5f);
}

__global__ void k_abs_spline(const float* __restrict__ in, float* __restrict__ out,
                             const float* __restrict__ c, int N)
{
  int i = blockIdx.x * blockDim.x + threadIdx.x;
  if (i >= N) return;
  out[i] = spline_eval(fabsf(in[i]), c, 31, 0.0f, 3.0f);
}

__global__ void k_spline3_clip(const float* __restrict__ in, float* __restrict__ out,
                               const float* __restrict__ c, const float* __restrict__ scaling,
                               int N)
{
  int i = blockIdx.x * blockDim.x + threadIdx.x;
  if (i >= N) return;
  int cch = (i >> 14) & (LCH - 1);
  int b   = i >> 21;
  float t = spline_eval(scaling[b * LCH + cch] * fabsf(in[i]), c, 31, 0.0f, 3.0f);
  out[i] = fminf(fmaxf(t, 0.01f), 1.0f);
}

__global__ void k_mask_mul(float* __restrict__ a, const float* __restrict__ m, int N)
{
  int i = blockIdx.x * blockDim.x + threadIdx.x;
  if (i >= N) return;
  float mm = m[i];
  a[i] *= mm * mm;
}

__global__ void k_combine(const float* __restrict__ x, const float* __restrict__ v,
                          float* __restrict__ out, const float* __restrict__ lmbda,
                          const float* __restrict__ inv1pl, int N)
{
  int i = blockIdx.x * blockDim.x + threadIdx.x;
  if (i >= N) return;
  int b = i >> 14;
  out[i] = (x[i] + lmbda[b] * v[i]) * inv1pl[b];
}

__global__ void k_resid_init(const float* __restrict__ y, const float* __restrict__ btb,
                             float* __restrict__ r, float* __restrict__ p,
                             const float* __restrict__ inv1pl, int N)
{
  int i = blockIdx.x * blockDim.x + threadIdx.x;
  if (i >= N) return;
  int b = i >> 14;
  float rv = y[i] * inv1pl[b] - btb[i];
  r[i] = rv;
  p[i] = rv;
}

__global__ void k_dot(const float* __restrict__ u, const float* __restrict__ v,
                      float* __restrict__ out)
{
  __shared__ float s[256];
  int b = blockIdx.x;
  const float* ub = u + (size_t)b * HW;
  const float* vb = v + (size_t)b * HW;
  float acc = 0.0f;
  for (int i = threadIdx.x; i < HW; i += 256) acc += ub[i] * vb[i];
  s[threadIdx.x] = acc;
  __syncthreads();
  for (int st = 128; st > 0; st >>= 1) {
    if (threadIdx.x < st) s[threadIdx.x] += s[threadIdx.x + st];
    __syncthreads();
  }
  if (threadIdx.x == 0) out[b] = s[0];
}

__global__ void k_alpha(const float* __restrict__ rn, const float* __restrict__ denom,
                        float* __restrict__ alpha)
{
  int t = threadIdx.x;
  if (t >= NB) return;
  bool act = rn[t] > 1e-6f;
  alpha[t] = act ? rn[t] / denom[t] : 0.0f;
}

__global__ void k_beta(float* __restrict__ rn, const float* __restrict__ rn2,
                       float* __restrict__ beta)
{
  int t = threadIdx.x;
  if (t >= NB) return;
  bool act = rn[t] > 1e-6f;
  beta[t] = act ? rn2[t] / rn[t] : 0.0f;
  rn[t] = rn2[t];
}

__global__ void k_xr(float* __restrict__ x, float* __restrict__ r,
                     const float* __restrict__ p, const float* __restrict__ Bp,
                     const float* __restrict__ alpha, int N)
{
  int i = blockIdx.x * blockDim.x + threadIdx.x;
  if (i >= N) return;
  int b = i >> 14;
  float al = alpha[b];
  x[i] += al * p[i];
  r[i] -= al * Bp[i];
}

__global__ void k_pupd(float* __restrict__ p, const float* __restrict__ r,
                       const float* __restrict__ beta, int N)
{
  int i = blockIdx.x * blockDim.x + threadIdx.x;
  if (i >= N) return;
  int b = i >> 14;
  p[i] = r[i] + beta[b] * p[i];
}

// ---------------------------------------------------------------------------
extern "C" void kernel_launch(void* const* d_in, const int* in_sizes, int n_in,
                              void* d_out, int out_size, void* d_ws, size_t ws_size,
                              hipStream_t stream)
{
  (void)in_sizes; (void)n_in; (void)out_size; (void)ws_size;
  const float* y     = (const float*)d_in[0];
  const float* sigma = (const float*)d_in[1];
  const float* w1_0  = (const float*)d_in[2];
  const float* w1_1  = (const float*)d_in[3];
  const float* w1_2  = (const float*)d_in[4];
  const float* m1_0  = (const float*)d_in[5];
  const float* m1_1  = (const float*)d_in[6];
  const float* m1_2  = (const float*)d_in[7];
  const float* m2_w  = (const float*)d_in[8];
  const float* m3_w  = (const float*)d_in[9];
  const float* c_sp1 = (const float*)d_in[10];
  const float* c_sp2 = (const float*)d_in[11];
  const float* c_sp3 = (const float*)d_in[12];
  const float* c_lam = (const float*)d_in[13];
  const float* c_scal= (const float*)d_in[14];
  // d_in[15]=n_out, d_in[16]=n_in are device-resident scalars; unreadable on
  // host during graph capture, so the setup values are baked in.
  constexpr int N_OUT = 2, N_ITER = 6;

  // ---- workspace carve (~200 MB) --------------------------------------
  char* basep = (char*)d_ws;
  size_t off = 0;
  auto carve = [&](size_t bytes) -> char* {
    char* cur = basep + off;
    off = (off + bytes + 255) & ~(size_t)255;
    return cur;
  };
  float* lmbda   = (float*)carve(NB * 4);
  float* inv1pl  = (float*)carve(NB * 4);
  float* scaling = (float*)carve(NB * LCH * 4);
  float* rn      = (float*)carve(NB * 4);
  float* rn2     = (float*)carve(NB * 4);
  float* denomv  = (float*)carve(NB * 4);
  float* alphav  = (float*)carve(NB * 4);
  float* betav   = (float*)carve(NB * 4);
  float* t4      = (float*)carve((size_t)NB * 4 * HW * 4);
  float* t8      = (float*)carve((size_t)NB * 8 * HW * 4);
  float* v1      = (float*)carve((size_t)NB * HW * 4);
  float* rbuf    = (float*)carve((size_t)NB * HW * 4);
  float* pbuf    = (float*)carve((size_t)NB * HW * 4);
  float* Bp      = (float*)carve((size_t)NB * HW * 4);
  float* abuf    = (float*)carve((size_t)NB * LCH * HW * 4);
  float* ubuf    = (float*)carve((size_t)NB * LCH * HW * 4);
  float* maskb   = (float*)carve((size_t)NB * LCH * HW * 4);
  bf16_t* w12p   = (bf16_t*)carve((size_t)128 * 672 * 2);
  bf16_t* w12tp  = (bf16_t*)carve((size_t)16 * 10368 * 2);
  bf16_t* m12p   = (bf16_t*)carve((size_t)128 * 672 * 2);
  bf16_t* m2p    = (bf16_t*)carve((size_t)128 * 1152 * 2);
  bf16_t* m3p    = (bf16_t*)carve((size_t)128 * 1152 * 2);
  float* w10p    = (float*)carve((size_t)4 * 81 * 4);
  float* w11p    = (float*)carve((size_t)8 * 324 * 4);
  float* w10tp   = (float*)carve((size_t)1 * 324 * 4);
  float* w11tp   = (float*)carve((size_t)4 * 648 * 4);
  float* m10p    = (float*)carve((size_t)4 * 81 * 4);
  float* m11p    = (float*)carve((size_t)8 * 324 * 4);

  float* X = (float*)d_out;                 // CG iterate, 8x1x128x128
  const int N1   = NB * HW;                 // 131072
  const int N128 = NB * LCH * HW;           // 16777216

  // ---- weight packing --------------------------------------------------
  auto packb = [&](const float* w, bf16_t* o, int O, int I, int K, int COP,
                   int RPAD, int flip, int chmin) {
    int total = COP * RPAD;
    k_pack<bf16_t><<<(total + 255) / 256, 256, 0, stream>>>(w, o, O, I, K, RPAD,
                                                            flip, chmin, total);
  };
  auto packf = [&](const float* w, float* o, int O, int I, int K, int COP,
                   int RPAD, int flip) {
    int total = COP * RPAD;
    k_pack<float><<<(total + 255) / 256, 256, 0, stream>>>(w, o, O, I, K, RPAD,
                                                           flip, 0, total);
  };
  packb(w1_2, w12p, 128,   8, 9, 128,   672, 0, 0);  // 8->128 9x9 (im2col path)
  packb(w1_2, w12tp,128,   8, 9,  16, 10368, 1, 1);  // adjoint 128->8 (chan-minor)
  packb(m1_2, m12p, 128,   8, 9, 128,   672, 0, 0);
  packb(m2_w, m2p,  128, 128, 3, 128,  1152, 0, 1);  // 128->128 3x3 (chan-minor)
  packb(m3_w, m3p,  128, 128, 3, 128,  1152, 0, 1);
  packf(w1_0, w10p,   4, 1, 9, 4,  81, 0);
  packf(w1_1, w11p,   8, 4, 9, 8, 324, 0);
  packf(w1_0, w10tp,  4, 1, 9, 1, 324, 1);           // adjoint 4->1
  packf(w1_1, w11tp,  8, 4, 9, 4, 648, 1);           // adjoint 8->4
  packf(m1_0, m10p,   4, 1, 9, 4,  81, 0);
  packf(m1_1, m11p,   8, 4, 9, 8, 324, 0);

  k_lam_scal<<<4, 256, 0, stream>>>(sigma, c_lam, c_scal, lmbda, inv1pl, scaling);
  hipMemsetAsync(X, 0, (size_t)N1 * 4, stream);      // c_k = 0

  const dim3 cgrid(WDIM / 16, HDIM, NB);
  auto conv_small_l = [&](const float* inp, const float* wpk, float* outp, int Ci, int Co) {
    int total = NB * Co * HW;
    k_conv_small<<<(total + 255) / 256, 256, 0, stream>>>(inp, wpk, outp, Ci, Co, 9, total);
  };

  // BtB(x) = (x + lambda * W1t(W1(x) * mask^2)) / (1 + lambda)
  auto btb = [&](const float* xin, float* bout) {
    conv_small_l(xin, w10p, t4, 1, 4);
    conv_small_l(t4,  w11p, t8, 4, 8);
    k_conv_wmma<8, 128, 128, 9, 0><<<cgrid, 256, 0, stream>>>(t8, w12p, abuf);
    k_mask_mul<<<N128 / 256, 256, 0, stream>>>(abuf, maskb, N128);
    k_conv_wmma<128, 16, 8, 9, 1><<<cgrid, 256, 0, stream>>>(abuf, w12tp, t8);
    conv_small_l(t8, w11tp, t4, 8, 4);
    conv_small_l(t4, w10tp, v1, 4, 1);
    k_combine<<<N1 / 256, 256, 0, stream>>>(xin, v1, bout, lmbda, inv1pl, N1);
  };

  for (int outer = 0; outer < N_OUT; ++outer) {
    // ---- mask = cal_mask(c_k) -----------------------------------------
    conv_small_l(X,  m10p, t4, 1, 4);
    conv_small_l(t4, m11p, t8, 4, 8);
    k_conv_wmma<8, 128, 128, 9, 0><<<cgrid, 256, 0, stream>>>(t8, m12p, abuf);
    k_abs_spline<<<N128 / 256, 256, 0, stream>>>(abuf, abuf, c_sp1, N128);
    k_conv_wmma<128, 128, 128, 3, 1><<<cgrid, 256, 0, stream>>>(abuf, m2p, ubuf);
    k_abs_spline<<<N128 / 256, 256, 0, stream>>>(ubuf, ubuf, c_sp2, N128);
    k_conv_wmma<128, 128, 128, 3, 1><<<cgrid, 256, 0, stream>>>(ubuf, m3p, maskb);
    k_spline3_clip<<<N128 / 256, 256, 0, stream>>>(maskb, maskb, c_sp3, scaling, N128);

    // ---- CG: x0 = X (previous c_k), b = y/(1+lambda) ------------------
    btb(X, Bp);
    k_resid_init<<<N1 / 256, 256, 0, stream>>>(y, Bp, rbuf, pbuf, inv1pl, N1);
    k_dot<<<NB, 256, 0, stream>>>(rbuf, rbuf, rn);
    for (int it = 0; it < N_ITER; ++it) {
      btb(pbuf, Bp);
      k_dot<<<NB, 256, 0, stream>>>(pbuf, Bp, denomv);
      k_alpha<<<1, NB, 0, stream>>>(rn, denomv, alphav);
      k_xr<<<N1 / 256, 256, 0, stream>>>(X, rbuf, pbuf, Bp, alphav, N1);
      k_dot<<<NB, 256, 0, stream>>>(rbuf, rbuf, rn2);
      k_beta<<<1, NB, 0, stream>>>(rn, rn2, betav);
      k_pupd<<<N1 / 256, 256, 0, stream>>>(pbuf, rbuf, betav, N1);
    }
  }
  // X (== d_out) now holds c_k.
}
